// SiameseNet_pos_att_cnn_adjust_41257455846139
// MI455X (gfx1250) — matrix operand order repaired and checked by reading, not compiled, not run on hardware
//
#include <hip/hip_runtime.h>
#include <hip/hip_bf16.h>
#include <math.h>

#define BB 512
#define LL 128
#define NN 64
#define HH 16
#define PWD 30
#define FEATN 480
#define MROWS (BB*LL)   // 65536

typedef _Float16 v16h __attribute__((ext_vector_type(16)));
typedef _Float16 v8h  __attribute__((ext_vector_type(8)));
typedef float    v8f  __attribute__((ext_vector_type(8)));

__device__ __forceinline__ float sigmoidf_(float x){ return 1.f/(1.f+__expf(-x)); }

// ---------------------------------------------------------------------------
// Attention: per block = one batch element b. score_x = softmax_L(x@(wxq@xk))*10,
// score_y = softmax_N((yk@wyq)@x)*10, out = x*sx[l]*sy[n] + pos[l,n].
// ---------------------------------------------------------------------------
__global__ void attention_kernel(const float* __restrict__ x,
                                 const float* __restrict__ wxq,
                                 const float* __restrict__ wxk,
                                 const float* __restrict__ wyq,
                                 const float* __restrict__ wyk,
                                 const float* __restrict__ pos,
                                 float* __restrict__ out)
{
    __shared__ float xs[LL*NN];
    __shared__ float xk[NN], tv[NN], sx[LL], yk[LL], uu[LL], sy[NN];
    __shared__ float red[LL];
    const int b = blockIdx.x, tid = threadIdx.x;      // 128 threads
    const float* xb = x + (size_t)b*LL*NN;
    for (int i = tid; i < LL*NN; i += 128) xs[i] = xb[i];
    __syncthreads();
    if (tid < NN) {                                    // xk[n] = sum_l x[l,n]*wxk[l]
        float s = 0.f;
        for (int l = 0; l < LL; ++l) s += xs[l*NN+tid]*wxk[l];
        xk[tid] = s;
    }
    __syncthreads();
    if (tid < NN) {                                    // tv = wxq @ xk
        float s = 0.f;
        for (int m = 0; m < NN; ++m) s += wxq[tid*NN+m]*xk[m];
        tv[tid] = s;
    }
    __syncthreads();
    {                                                  // sx[l] = x[l,:]·tv ; softmax over L
        float s = 0.f;
        for (int n = 0; n < NN; ++n) s += xs[tid*NN+n]*tv[n];
        sx[tid] = s; red[tid] = s;
    }
    __syncthreads();
    for (int st = 64; st > 0; st >>= 1) { if (tid < st) red[tid] = fmaxf(red[tid], red[tid+st]); __syncthreads(); }
    float mx = red[0];
    __syncthreads();
    float ex = __expf(sx[tid]-mx);
    red[tid] = ex;
    __syncthreads();
    for (int st = 64; st > 0; st >>= 1) { if (tid < st) red[tid] += red[tid+st]; __syncthreads(); }
    float denom = red[0];
    __syncthreads();
    sx[tid] = ex/denom*10.f;
    {                                                  // yk[l] = wyk·x[l,:]
        float s = 0.f;
        for (int n = 0; n < NN; ++n) s += wyk[n]*xs[tid*NN+n];
        yk[tid] = s;
    }
    __syncthreads();
    {                                                  // uu[m] = sum_l yk[l]*wyq[l,m]
        float s = 0.f;
        for (int l = 0; l < LL; ++l) s += yk[l]*wyq[l*LL+tid];
        uu[tid] = s;
    }
    __syncthreads();
    float yv = 0.f;
    if (tid < NN) {                                    // y_[n] = sum_m uu[m]*x[m,n]
        float s = 0.f;
        for (int m = 0; m < LL; ++m) s += uu[m]*xs[m*NN+tid];
        yv = s;
    }
    red[tid] = (tid < NN) ? yv : -1e30f;
    __syncthreads();
    for (int st = 64; st > 0; st >>= 1) { if (tid < st) red[tid] = fmaxf(red[tid], red[tid+st]); __syncthreads(); }
    mx = red[0];
    __syncthreads();
    ex = (tid < NN) ? __expf(yv-mx) : 0.f;
    red[tid] = ex;
    __syncthreads();
    for (int st = 64; st > 0; st >>= 1) { if (tid < st) red[tid] += red[tid+st]; __syncthreads(); }
    denom = red[0];
    __syncthreads();
    if (tid < NN) sy[tid] = ex/denom*10.f;
    __syncthreads();
    float* ob = out + (size_t)b*LL*NN;
    for (int i = tid; i < LL*NN; i += 128) {
        int l = i >> 6, n = i & 63;
        ob[i] = xs[i]*sx[l]*sy[n] + pos[i];
    }
}

// ---------------------------------------------------------------------------
// Conv(1x4)+ReLU -> BN affine -> maxpool(1,2); emit f16 features [row, h*30+p].
// One block per (b,l) row.
// ---------------------------------------------------------------------------
__global__ void conv_bn_pool(const float* __restrict__ att,
                             const float* __restrict__ cw, const float* __restrict__ cb,
                             const float* __restrict__ gamma, const float* __restrict__ beta,
                             const float* __restrict__ mean, const float* __restrict__ var,
                             _Float16* __restrict__ feat)
{
    __shared__ float row[NN];
    const int rowIdx = blockIdx.x, tid = threadIdx.x;  // 128 threads
    const float* src = att + (size_t)rowIdx*NN;
    if (tid < NN) row[tid] = src[tid];
    __syncthreads();
    for (int idx = tid; idx < HH*PWD; idx += 128) {
        int h = idx / PWD, p = idx % PWD;
        float w0 = cw[h*4+0], w1 = cw[h*4+1], w2 = cw[h*4+2], w3 = cw[h*4+3];
        float bv = cb[h];
        float scale = gamma[h]*rsqrtf(var[h]+1e-5f);
        float mh = mean[h], bt = beta[h];
        int j = 2*p;
        float z0 = fmaxf(0.f, bv + w0*row[j]   + w1*row[j+1] + w2*row[j+2] + w3*row[j+3]);
        float z1 = fmaxf(0.f, bv + w0*row[j+1] + w1*row[j+2] + w2*row[j+3] + w3*row[j+4]);
        z0 = (z0-mh)*scale+bt;
        z1 = (z1-mh)*scale+bt;
        feat[(size_t)rowIdx*FEATN + idx] = (_Float16)fmaxf(z0, z1);
    }
}

// ---------------------------------------------------------------------------
// WMMA GEMM: C[M, nReal] = A[M,K](f16,row-major) @ W[nPad,K](f16,row-major).T
//            + bias, optional leaky-relu; out f32 or f16. K multiple of 32.
// One wave per 16x16 tile; fragments packed per CDNA5 ISA VGPR layouts.
// ---------------------------------------------------------------------------
__global__ void wmma_gemm(const _Float16* __restrict__ A, int lda,
                          const _Float16* __restrict__ W, int ldw,
                          const float* __restrict__ bias,
                          float* __restrict__ outF, _Float16* __restrict__ outH,
                          int ldc, int K, int nTiles, int nReal, int act)
{
    const int lane  = threadIdx.x & 31;
    const int wave  = threadIdx.x >> 5;                 // 4 waves/block
    const int mtile = blockIdx.x;
    const int ntile = blockIdx.y*4 + wave;
    if (ntile >= nTiles) return;                        // uniform per wave
    const int hsel = lane >> 4;                         // 0: lanes 0-15, 1: 16-31
    const int mrow = (mtile << 4) + (lane & 15);
    const int ncol = (ntile << 4) + (lane & 15);
    // A lane layout: halves 0..7 -> K=base..base+7, halves 8..15 -> K=base+16..23, base=8*hsel
    const _Float16* Ap = A + (size_t)mrow*lda + (hsel ? 8 : 0);
    // B lane layout: halves 0..7 -> K=base..base+7, halves 8..15 -> K=base+16..23, base=8*hsel
    const _Float16* Wp = W + (size_t)ncol*ldw + (hsel ? 8 : 0);
    v8f acc = {0.f,0.f,0.f,0.f,0.f,0.f,0.f,0.f};
    for (int k0 = 0; k0 < K; k0 += 32) {
        v8h a0 = *(const v8h*)(Ap + k0);
        v8h a1 = *(const v8h*)(Ap + k0 + 16);
        v8h b0 = *(const v8h*)(Wp + k0);
        v8h b1 = *(const v8h*)(Wp + k0 + 16);
        if (k0 + 32 < K) {
            __builtin_prefetch(Ap + k0 + 32, 0, 1);     // -> global_prefetch_b8
            __builtin_prefetch(Wp + k0 + 32, 0, 1);
        }
        v16h av = __builtin_shufflevector(a0, a1, 0,1,2,3,4,5,6,7,8,9,10,11,12,13,14,15);
        v16h bv = __builtin_shufflevector(b0, b1, 0,1,2,3,4,5,6,7,8,9,10,11,12,13,14,15);
        acc = __builtin_amdgcn_wmma_f32_16x16x32_f16(false, av, false, bv,
                                                     (short)0, acc, false, false);
    }
    if (ncol < nReal) {
        const float bz = bias ? bias[ncol] : 0.f;
        const int mbase = (mtile << 4) + (hsel << 3);   // VGPR r -> M = 8*hsel + r
        #pragma unroll
        for (int r = 0; r < 8; ++r) {
            float v = acc[r] + bz;
            if (act == 1) v = (v > 0.f) ? v : 0.01f*v;
            size_t o = (size_t)(mbase + r)*ldc + ncol;
            if (outF) outF[o] = v; else outH[o] = (_Float16)v;
        }
    }
}

// ---------------------------------------------------------------------------
// LSTM recurrence: 512 sequential steps; rows independent -> block per row r.
// 64 threads, one gate column each; h/c state + gate exchange in LDS.
// mode 0: write h1 as f16 [t*128+r][32] (upper 16 zero-padded for K=32 GEMM)
// mode 1: write h2 as f16 [t][r*16+k]   (fc input layout, ld=2048)
// ---------------------------------------------------------------------------
__global__ void lstm_scan(const float* __restrict__ gin, const float* __restrict__ Whh,
                          _Float16* __restrict__ hOut, int mode)
{
    __shared__ float sW[64*16];
    __shared__ float sh[16], sc[16], sg[64];
    const int tid = threadIdx.x;    // 64
    const int r   = blockIdx.x;     // 0..127
    for (int i = tid; i < 64*16; i += 64) sW[i] = Whh[i];
    if (tid < 16) { sh[tid] = 0.f; sc[tid] = 0.f; }
    __syncthreads();
    for (int t = 0; t < BB; ++t) {
        float g = gin[((size_t)t*LL + r)*64 + tid];
        #pragma unroll
        for (int k = 0; k < 16; ++k) g += sh[k]*sW[tid*16 + k];
        sg[tid] = g;
        __syncthreads();
        if (tid < 16) {
            float gi = sg[tid], gf = sg[16+tid], gg = sg[32+tid], go = sg[48+tid];
            float c = sigmoidf_(gf)*sc[tid] + sigmoidf_(gi)*tanhf(gg);
            float h = sigmoidf_(go)*tanhf(c);
            sc[tid] = c; sh[tid] = h;
            if (mode == 0) {
                _Float16* rp = hOut + ((size_t)t*LL + r)*32;
                rp[tid] = (_Float16)h;
                rp[16+tid] = (_Float16)0.f;
            } else {
                hOut[(size_t)t*2048 + r*16 + tid] = (_Float16)h;
            }
        }
        __syncthreads();
    }
}

// f32 -> f16 convert with zero padding: dst [dR,dC], src [sR,sC]
__global__ void cvt_pad_f16(const float* __restrict__ src, _Float16* __restrict__ dst,
                            int sR, int sC, int dR, int dC)
{
    int idx = blockIdx.x*blockDim.x + threadIdx.x;
    if (idx >= dR*dC) return;
    int rr = idx / dC, cc = idx % dC;
    dst[idx] = (rr < sR && cc < sC) ? (_Float16)src[rr*sC + cc] : (_Float16)0.f;
}

__global__ void add_vec(const float* __restrict__ a, const float* __restrict__ b,
                        float* __restrict__ o, int n)
{
    int i = blockIdx.x*blockDim.x + threadIdx.x;
    if (i < n) o[i] = a[i] + b[i];
}

extern "C" void kernel_launch(void* const* d_in, const int* in_sizes, int n_in,
                              void* d_out, int out_size, void* d_ws, size_t ws_size,
                              hipStream_t stream) {
    (void)in_sizes; (void)n_in; (void)out_size; (void)ws_size;
    const float* in0[2] = { (const float*)d_in[0], (const float*)d_in[1] };
    const float* wxq  = (const float*)d_in[2];
    const float* wxk  = (const float*)d_in[3];
    const float* wyq  = (const float*)d_in[4];
    const float* wyk  = (const float*)d_in[5];
    const float* pos  = (const float*)d_in[6];
    const float* cw   = (const float*)d_in[7];
    const float* cb   = (const float*)d_in[8];
    const float* gma  = (const float*)d_in[9];
    const float* bta  = (const float*)d_in[10];
    const float* mn   = (const float*)d_in[11];
    const float* vr   = (const float*)d_in[12];
    const float* Wih0 = (const float*)d_in[13];
    const float* Whh0 = (const float*)d_in[14];
    const float* bih0 = (const float*)d_in[15];
    const float* bhh0 = (const float*)d_in[16];
    const float* Wih1 = (const float*)d_in[17];
    const float* Whh1 = (const float*)d_in[18];
    const float* bih1 = (const float*)d_in[19];
    const float* bhh1 = (const float*)d_in[20];
    const float* fc1w = (const float*)d_in[21];
    const float* fc1b = (const float*)d_in[22];
    const float* fc2w = (const float*)d_in[23];
    const float* fc2b = (const float*)d_in[24];
    const float* fc3w = (const float*)d_in[25];
    const float* fc3b = (const float*)d_in[26];
    float* outp = (float*)d_out;

    char* ws = (char*)d_ws;
    size_t off = 0;
    auto alloc = [&](size_t bytes) -> char* {
        char* p = ws + off; off = (off + bytes + 255) & ~(size_t)255; return p;
    };
    _Float16* wih0h = (_Float16*)alloc((size_t)64*480*2);
    _Float16* wih1h = (_Float16*)alloc((size_t)64*32*2);     // padded K 16->32
    _Float16* fc1wh = (_Float16*)alloc((size_t)1024*2048*2);
    _Float16* fc2wh = (_Float16*)alloc((size_t)1024*1024*2);
    _Float16* fc3wh = (_Float16*)alloc((size_t)32*1024*2);   // padded N 20->32
    float*    bias0 = (float*)alloc(64*4);
    float*    bias1 = (float*)alloc(64*4);
    float*    att   = (float*)alloc((size_t)MROWS*NN*4);
    _Float16* feat  = (_Float16*)alloc((size_t)MROWS*FEATN*2);
    float*    gin   = (float*)alloc((size_t)MROWS*64*4);     // reused layer0/layer1
    _Float16* h1    = (_Float16*)alloc((size_t)MROWS*32*2);
    _Float16* h2    = (_Float16*)alloc((size_t)BB*2048*2);
    _Float16* f1    = (_Float16*)alloc((size_t)BB*1024*2);
    _Float16* f2    = (_Float16*)alloc((size_t)BB*1024*2);

    // --- one-time weight prep (f32 -> f16, zero-padded where shapes are odd) ---
    auto cvt = [&](const float* s, _Float16* d, int sR, int sC, int dR, int dC) {
        int total = dR*dC;
        cvt_pad_f16<<<(total + 255)/256, 256, 0, stream>>>(s, d, sR, sC, dR, dC);
    };
    cvt(Wih0, wih0h, 64, 480, 64, 480);
    cvt(Wih1, wih1h, 64, 16,  64, 32);
    cvt(fc1w, fc1wh, 1024, 2048, 1024, 2048);
    cvt(fc2w, fc2wh, 1024, 1024, 1024, 1024);
    cvt(fc3w, fc3wh, 20, 1024, 32, 1024);
    add_vec<<<1, 64, 0, stream>>>(bih0, bhh0, bias0, 64);
    add_vec<<<1, 64, 0, stream>>>(bih1, bhh1, bias1, 64);

    // --- per-input pipeline (two independent Siamese branches) ---
    for (int i = 0; i < 2; ++i) {
        attention_kernel<<<BB, 128, 0, stream>>>(in0[i], wxq, wxk, wyq, wyk, pos, att);
        conv_bn_pool<<<MROWS, 128, 0, stream>>>(att, cw, cb, gma, bta, mn, vr, feat);
        // LSTM layer0 input projection: [65536,480]@[480,64] -> gin (f32)
        wmma_gemm<<<dim3(MROWS/16, 1), 128, 0, stream>>>(
            feat, FEATN, wih0h, FEATN, bias0, gin, nullptr, 64, FEATN, 4, 64, 0);
        lstm_scan<<<LL, 64, 0, stream>>>(gin, Whh0, h1, 0);
        // LSTM layer1 input projection: [65536,32pad]@[32pad,64] -> gin (f32)
        wmma_gemm<<<dim3(MROWS/16, 1), 128, 0, stream>>>(
            h1, 32, wih1h, 32, bias1, gin, nullptr, 64, 32, 4, 64, 0);
        lstm_scan<<<LL, 64, 0, stream>>>(gin, Whh1, h2, 1);
        // fc1: [512,2048]@[2048,1024] + leaky -> f16
        wmma_gemm<<<dim3(BB/16, 16), 128, 0, stream>>>(
            h2, 2048, fc1wh, 2048, fc1b, nullptr, f1, 1024, 2048, 64, 1024, 1);
        // fc2: [512,1024]@[1024,1024] + leaky -> f16
        wmma_gemm<<<dim3(BB/16, 16), 128, 0, stream>>>(
            f1, 1024, fc2wh, 1024, fc2b, nullptr, f2, 1024, 1024, 64, 1024, 1);
        // fc3: [512,1024]@[1024,20(pad32)] -> f32 output slice
        wmma_gemm<<<dim3(BB/16, 1), 128, 0, stream>>>(
            f2, 1024, fc3wh, 1024, fc3b, outp + (size_t)i*BB*20, nullptr,
            20, 1024, 2, 20, 0);
    }
}
